// MLAFromNotebook_67662914781410
// MI455X (gfx1250) — matrix operand order repaired and checked by reading, not compiled, not run on hardware
//
#include <hip/hip_runtime.h>

typedef __bf16 bf16_t;
typedef __attribute__((ext_vector_type(16))) __bf16 v16bf;
typedef __attribute__((ext_vector_type(8)))  __bf16 v8bf;
typedef __attribute__((ext_vector_type(4)))  __bf16 v4bf;
typedef __attribute__((ext_vector_type(8)))  float  v8f;
typedef __attribute__((ext_vector_type(4)))  unsigned v4u;
typedef __attribute__((ext_vector_type(8)))  int      v8i;
typedef __attribute__((ext_vector_type(4)))  int      v4i;

#define D_MODEL   1024
#define N_HEADS   16
#define HEAD_DIM  64
#define KV_LATENT 256
#define SEQ       2048
#define BATCH     2
#define M_TOT     (BATCH * SEQ)   // 4096

#if __has_builtin(__builtin_amdgcn_tensor_load_to_lds) && \
    __has_builtin(__builtin_amdgcn_s_wait_tensorcnt)
#define USE_TDM 1
#else
#define USE_TDM 0
#endif

// ---------------------------------------------------------------------------
// helpers
// ---------------------------------------------------------------------------
__device__ __forceinline__ v16bf ld2x8(const bf16_t* p0, const bf16_t* p1) {
    union { v16bf v; v8bf h[2]; } u;
    u.h[0] = *reinterpret_cast<const v8bf*>(p0);
    u.h[1] = *reinterpret_cast<const v8bf*>(p1);
    return u.v;
}

__device__ __forceinline__ v8f wmma_bf16(v16bf a, v16bf b, v8f c) {
    // D = A(16x32 bf16) * B(32x16 bf16) + C(16x16 f32)
    return __builtin_amdgcn_wmma_f32_16x16x32_bf16(false, a, false, b,
                                                   (short)0, c, false, false);
}

#if USE_TDM
// TDM: load an 8-row x 32-col bf16 tile (row stride K elems) from global into
// LDS at lds_off, inserting 16B of padding after every 64B row (-> 80B stride).
// 6-arg builtin variant: (v4u g0, v8i g1, v4i g2, v4i g3, v8i g4, i32 cpol)
__device__ __forceinline__ void tdm_load_tile8x32(unsigned lds_off,
                                                  const bf16_t* gaddr, int K) {
    unsigned long long ga = (unsigned long long)(uintptr_t)gaddr;
    unsigned ga_lo = __builtin_amdgcn_readfirstlane((unsigned)(ga & 0xFFFFFFFFu));
    unsigned ga_hi = __builtin_amdgcn_readfirstlane((unsigned)(ga >> 32));
    unsigned ldso  = __builtin_amdgcn_readfirstlane(lds_off);
    // D# group 0: count=1 | lds_addr | global_addr[56:0] | type=2
    v4u g0 = { 1u, ldso, ga_lo, (ga_hi & 0x01FFFFFFu) | (2u << 30) };
    // D# group 1: data_size=2B, pad_enable, pad_interval=16DW, pad_amount=4DW,
    // tensor_dim0=32, tensor_dim1=8, tile_dim0=32, tile_dim1=8, dim0_stride=K
    v8i g1 = { (int)((1u << 16) | (1u << 20) | (3u << 22) | (3u << 25)),
               (int)(32u << 16),          // tensor_dim0[15:0] @ bits 63:48
               (int)(8u  << 16),          // tensor_dim1[15:0] @ bits 95:80
               (int)(32u << 16),          // tile_dim0 @ bits 127:112
               (int)(8u),                 // tile_dim1
               (int)(K),                  // tensor_dim0_stride[31:0]
               0, 0 };
    v4i z4 = {0, 0, 0, 0};
    v8i z8 = {0, 0, 0, 0, 0, 0, 0, 0};
    __builtin_amdgcn_tensor_load_to_lds(g0, g1, z4, z4, z8, 0);
}
#endif

// ---------------------------------------------------------------------------
// elementwise conversion kernels
// ---------------------------------------------------------------------------
__global__ void cvt_f32_to_bf16(const float* __restrict__ in,
                                bf16_t* __restrict__ out, int n4) {
    int i = blockIdx.x * blockDim.x + threadIdx.x;
    if (i >= n4) return;
    float4 f = reinterpret_cast<const float4*>(in)[i];
    v4bf o;
    o.x = (bf16_t)f.x; o.y = (bf16_t)f.y; o.z = (bf16_t)f.z; o.w = (bf16_t)f.w;
    reinterpret_cast<v4bf*>(out)[i] = o;
}

// WT[n*K + k] = (bf16) W[k*N + n]   (store weight transposed, N-major)
__global__ void transpose_to_bf16(const float* __restrict__ W,
                                  bf16_t* __restrict__ WT, int K, int N) {
    int idx = blockIdx.x * blockDim.x + threadIdx.x;
    if (idx >= N * K) return;
    int n = idx / K;
    int k = idx - n * K;
    WT[idx] = (bf16_t)W[(size_t)k * N + n];
}

// ---------------------------------------------------------------------------
// LayerNorm over 256-wide latent rows: one wave32 per row, 8 elems / lane
// ---------------------------------------------------------------------------
__global__ void ln_rows(const float* __restrict__ lat,
                        const float* __restrict__ gamma,
                        const float* __restrict__ beta,
                        bf16_t* __restrict__ out) {
    int gid  = blockIdx.x * blockDim.x + threadIdx.x;
    int row  = gid >> 5;
    int lane = gid & 31;
    const float* rp = lat + (size_t)row * KV_LATENT + lane * 8;
    float4 a = reinterpret_cast<const float4*>(rp)[0];
    float4 b = reinterpret_cast<const float4*>(rp)[1];
    float x[8] = {a.x, a.y, a.z, a.w, b.x, b.y, b.z, b.w};
    float s = 0.f, q = 0.f;
#pragma unroll
    for (int i = 0; i < 8; ++i) { s += x[i]; q += x[i] * x[i]; }
#pragma unroll
    for (int off = 1; off < 32; off <<= 1) {
        s += __shfl_xor(s, off, 32);
        q += __shfl_xor(q, off, 32);
    }
    float mu  = s * (1.f / 256.f);
    float var = q * (1.f / 256.f) - mu * mu;
    float rs  = rsqrtf(var + 1e-5f);
    union { v8bf v; bf16_t e[8]; } o;
#pragma unroll
    for (int i = 0; i < 8; ++i) {
        float g  = gamma[lane * 8 + i];
        float be = beta[lane * 8 + i];
        o.e[i] = (bf16_t)((x[i] - mu) * rs * g + be);
    }
    *reinterpret_cast<v8bf*>(out + (size_t)row * KV_LATENT + lane * 8) = o.v;
}

// ---------------------------------------------------------------------------
// Tiled bf16 WMMA GEMM:  C(MxN) = A(MxK) * BT(NxK)^T
//   block = 256 threads (8 waves); block tile = 128M x 64N; wave tile = 16M x 64N
//   B K-slice (64N x 32K) double-buffered in LDS via TDM: the DMA for slice
//   i+1 is in flight while WMMAs consume slice i (TDM ops complete in-order
//   per wave, so s_wait_tensorcnt 0 at the top covers the prefetched tile).
// Epilogues: 0 = f32 store, 1 = RoPE+scale -> bf16 per-head [bh][t][hd],
//            2 = transposed bf16 store -> vt [bh][hd][t]
// ---------------------------------------------------------------------------
template <int EPI>
__global__ void gemm_wmma(const bf16_t* __restrict__ A,
                          const bf16_t* __restrict__ BT,
                          void* __restrict__ out,
                          int N, int K, float scale) {
    int tid  = threadIdx.x;
    int lane = tid & 31;
    int l15  = lane & 15;
    int half = lane >> 4;
    int wave = tid >> 5;
    int n0   = blockIdx.x * 64;
    int m0   = blockIdx.y * 128 + wave * 16;

    __shared__ __align__(16) bf16_t Bs[2][64][40];  // 2 x (64 cols x 32 K), 80B stride

    v8f c[4];
#pragma unroll
    for (int j = 0; j < 4; ++j) c[j] = (v8f){0.f,0.f,0.f,0.f,0.f,0.f,0.f,0.f};

    const bf16_t* arow = A + (size_t)(m0 + l15) * K;
#if USE_TDM
    unsigned lds0 = (unsigned)(uintptr_t)&Bs[0][0][0] + (unsigned)(wave * 8 * 80);
    unsigned lds1 = (unsigned)(uintptr_t)&Bs[1][0][0] + (unsigned)(wave * 8 * 80);
    const bf16_t* bwave = BT + (size_t)(n0 + wave * 8) * K;
    tdm_load_tile8x32(lds0, bwave, K);              // prime buffer 0
#else
    int ncol = tid >> 2;          // 0..63
    int part = tid & 3;           // 0..3 (8 bf16 each)
    const bf16_t* bsrc = BT + (size_t)(n0 + ncol) * K + part * 8;
#endif

    int p = 0;
    for (int k0 = 0; k0 < K; k0 += 32) {
#if USE_TDM
        __builtin_amdgcn_s_wait_tensorcnt(0);       // slice k0 resident (in-order TDM)
        __syncthreads();                            // ... in every wave's rows
        if (k0 + 32 < K)                            // overlap DMA of next slice
            tdm_load_tile8x32(p ? lds0 : lds1, bwave + k0 + 32, K);
#else
        *reinterpret_cast<v8bf*>(&Bs[0][ncol][part * 8]) =
            *reinterpret_cast<const v8bf*>(bsrc + k0);
        if (k0 + 32 < K)
            __builtin_prefetch(bsrc + k0 + 32, 0, 0);
        __syncthreads();
#endif

        v16bf a = ld2x8(arow + k0 + 16 * half, arow + k0 + 16 * half + 8);
#pragma unroll
        for (int j = 0; j < 4; ++j) {
            const bf16_t* bp = &Bs[USE_TDM ? p : 0][16 * j + l15][16 * half];
            v16bf b = ld2x8(bp, bp + 8);
            c[j] = wmma_bf16(a, b, c[j]);
        }
        __syncthreads();                            // all reads of slice done
        p ^= 1;
    }

    if constexpr (EPI == 0) {                       // plain f32 store
        float* C = (float*)out;
#pragma unroll
        for (int j = 0; j < 4; ++j)
#pragma unroll
            for (int r = 0; r < 8; ++r) {
                int m = m0 + r + 8 * half;
                C[(size_t)m * N + n0 + 16 * j + l15] = c[j][r];
            }
    } else if constexpr (EPI == 1) {                // RoPE + scale -> [bh][t][hd]
        bf16_t* dst = (bf16_t*)out;
        int h = n0 >> 6;                            // one head per 64-wide wave tile
#pragma unroll
        for (int jj = 0; jj < 2; ++jj) {
            int i = 16 * jj + l15;                  // rotary freq index 0..31
            float invf = __expf(-(float)i * (9.210340372f / 32.f)); // 10000^(-i/32)
#pragma unroll
            for (int r = 0; r < 8; ++r) {
                int m = m0 + r + 8 * half;
                int b = m >> 11;
                int t = m & (SEQ - 1);
                float ang = (float)t * invf;
                float cs = __cosf(ang), sn = __sinf(ang);
                float x1 = c[jj][r], x2 = c[jj + 2][r];
                float o1 = (x1 * cs - x2 * sn) * scale;
                float o2 = (x1 * sn + x2 * cs) * scale;
                bf16_t* base = dst + (((size_t)b * N_HEADS + h) * SEQ + t) * HEAD_DIM;
                base[16 * jj + l15]      = (bf16_t)o1;
                base[32 + 16 * jj + l15] = (bf16_t)o2;
            }
        }
    } else {                                        // EPI == 2: V transposed [bh][hd][t]
        bf16_t* dst = (bf16_t*)out;
        int h = n0 >> 6;
#pragma unroll
        for (int j = 0; j < 4; ++j) {
            int hd = 16 * j + l15;
#pragma unroll
            for (int r = 0; r < 8; ++r) {
                int m = m0 + r + 8 * half;
                int b = m >> 11;
                int t = m & (SEQ - 1);
                dst[(((size_t)b * N_HEADS + h) * HEAD_DIM + hd) * SEQ + t] = (bf16_t)c[j][r];
            }
        }
    }
}

// ---------------------------------------------------------------------------
// Flash-style causal attention. One wave per 16-row Q tile, 32-key blocks
// (keeps S/P state at 16 VGPRs -> no scratch spills).
// q pre-scaled by 1/sqrt(64). Output -> attn_out bf16 (b,t,d) layout.
// ---------------------------------------------------------------------------
__global__ void mla_attention(const bf16_t* __restrict__ qh,
                              const bf16_t* __restrict__ kh,
                              const bf16_t* __restrict__ vt,
                              bf16_t* __restrict__ outb) {
    int tid  = threadIdx.x;
    int lane = tid & 31;
    int l15  = lane & 15;
    int half = lane >> 4;
    int wave = tid >> 5;
    int bh   = blockIdx.y;
    int m0   = blockIdx.x * 128 + wave * 16;

    __shared__ __align__(16) bf16_t P[8][16][40];   // per-wave 16x32 P staging

    const bf16_t* qbase = qh + (size_t)bh * SEQ * HEAD_DIM;
    const bf16_t* kbase = kh + (size_t)bh * SEQ * HEAD_DIM;
    const bf16_t* vbase = vt + (size_t)bh * HEAD_DIM * SEQ;

    // Q fragments for both 32-deep K-steps of head_dim=64
    const bf16_t* qrow = qbase + (size_t)(m0 + l15) * HEAD_DIM;
    v16bf aq0 = ld2x8(qrow + 16 * half,      qrow + 16 * half + 8);
    v16bf aq1 = ld2x8(qrow + 32 + 16 * half, qrow + 32 + 16 * half + 8);

    float mrow[8], lrow[8];
    v8f o[4];
#pragma unroll
    for (int r = 0; r < 8; ++r) { mrow[r] = -1e30f; lrow[r] = 0.f; }
#pragma unroll
    for (int j = 0; j < 4; ++j) o[j] = (v8f){0.f,0.f,0.f,0.f,0.f,0.f,0.f,0.f};

    int rend = m0 + 15;
    for (int t0 = 0; t0 <= rend; t0 += 32) {
        // ---- S = Q * K^T for 32 keys (2 x 16-wide tiles) ----
        v8f s[2];
#pragma unroll
        for (int j = 0; j < 2; ++j) {
            s[j] = (v8f){0.f,0.f,0.f,0.f,0.f,0.f,0.f,0.f};
            int key = t0 + 16 * j + l15;
            int kc  = key < SEQ - 1 ? key : SEQ - 1;      // clamp OOB (masked anyway)
            const bf16_t* kr = kbase + (size_t)kc * HEAD_DIM;
            v16bf b0 = ld2x8(kr + 16 * half,      kr + 16 * half + 8);
            v16bf b1 = ld2x8(kr + 32 + 16 * half, kr + 32 + 16 * half + 8);
            s[j] = wmma_bf16(aq0, b0, s[j]);
            s[j] = wmma_bf16(aq1, b1, s[j]);
        }
        // ---- causal mask on partial blocks ----
        if (t0 + 31 > m0) {
#pragma unroll
            for (int j = 0; j < 2; ++j) {
                int key = t0 + 16 * j + l15;
#pragma unroll
                for (int r = 0; r < 8; ++r) {
                    int row = m0 + r + 8 * half;
                    if (key > row) s[j][r] = -1e30f;
                }
            }
        }
        // ---- online softmax (row reductions across 16-lane groups) ----
#pragma unroll
        for (int r = 0; r < 8; ++r) {
            float tm = fmaxf(s[0][r], s[1][r]);
#pragma unroll
            for (int off = 1; off < 16; off <<= 1)
                tm = fmaxf(tm, __shfl_xor(tm, off, 32));
            float mn = fmaxf(mrow[r], tm);
            float sc = __expf(mrow[r] - mn);
            mrow[r] = mn;
            float p0 = __expf(s[0][r] - mn);
            float p1 = __expf(s[1][r] - mn);
            s[0][r] = p0;
            s[1][r] = p1;
            float rs = p0 + p1;
#pragma unroll
            for (int off = 1; off < 16; off <<= 1)
                rs += __shfl_xor(rs, off, 32);
            lrow[r] = lrow[r] * sc + rs;
#pragma unroll
            for (int jo = 0; jo < 4; ++jo) o[jo][r] *= sc;
        }
        // ---- re-stripe P: C-layout -> A-layout via LDS ----
#pragma unroll
        for (int j = 0; j < 2; ++j)
#pragma unroll
            for (int r = 0; r < 8; ++r)
                P[wave][r + 8 * half][16 * j + l15] = (bf16_t)s[j][r];
        asm volatile("s_wait_dscnt 0" ::: "memory");

        const bf16_t* prow = &P[wave][l15][0];
        v16bf ap = ld2x8(prow + 16 * half, prow + 16 * half + 8);

        // ---- O += P * V ----
        int bt = t0 + 16 * half;
        if (bt > SEQ - 16) bt = SEQ - 16;
#pragma unroll
        for (int jo = 0; jo < 4; ++jo) {
            int hd = 16 * jo + l15;
            const bf16_t* vcol = vbase + (size_t)hd * SEQ;
            v16bf bv = ld2x8(vcol + bt, vcol + bt + 8);
            o[jo] = wmma_bf16(ap, bv, o[jo]);
        }
        asm volatile("s_wait_dscnt 0" ::: "memory");
    }

    // ---- normalize + store to (b, t, d) bf16 ----
    int b = bh >> 4, h = bh & 15;
#pragma unroll
    for (int jo = 0; jo < 4; ++jo) {
#pragma unroll
        for (int r = 0; r < 8; ++r) {
            int t = m0 + r + 8 * half;
            float v = o[jo][r] / lrow[r];
            outb[((size_t)(b * SEQ + t)) * D_MODEL + h * HEAD_DIM + 16 * jo + l15] =
                (bf16_t)v;
        }
    }
}

// ---------------------------------------------------------------------------
// launch
// ---------------------------------------------------------------------------
extern "C" void kernel_launch(void* const* d_in, const int* in_sizes, int n_in,
                              void* d_out, int out_size, void* d_ws, size_t ws_size,
                              hipStream_t stream) {
    (void)in_sizes; (void)n_in; (void)out_size; (void)ws_size;
    const float* x        = (const float*)d_in[0];
    const float* W_q      = (const float*)d_in[1];
    const float* W_dkv    = (const float*)d_in[2];
    const float* W_uk     = (const float*)d_in[3];
    const float* W_uv     = (const float*)d_in[4];
    const float* W_o      = (const float*)d_in[5];
    const float* ln_gamma = (const float*)d_in[6];
    const float* ln_beta  = (const float*)d_in[7];

    char* p = (char*)d_ws;
    auto alloc = [&](size_t bytes) -> void* {
        void* r = (void*)p;
        p += (bytes + 255) & ~(size_t)255;
        return r;
    };
    bf16_t* xb      = (bf16_t*)alloc((size_t)M_TOT * D_MODEL * 2);
    bf16_t* WqT     = (bf16_t*)alloc((size_t)D_MODEL * D_MODEL * 2);
    bf16_t* WdkvT   = (bf16_t*)alloc((size_t)KV_LATENT * D_MODEL * 2);
    bf16_t* WukT    = (bf16_t*)alloc((size_t)D_MODEL * KV_LATENT * 2);
    bf16_t* WuvT    = (bf16_t*)alloc((size_t)D_MODEL * KV_LATENT * 2);
    bf16_t* WoT     = (bf16_t*)alloc((size_t)D_MODEL * D_MODEL * 2);
    float*  lat_f32 = (float*) alloc((size_t)M_TOT * KV_LATENT * 4);
    bf16_t* latn    = (bf16_t*)alloc((size_t)M_TOT * KV_LATENT * 2);
    bf16_t* q_h     = (bf16_t*)alloc((size_t)BATCH * N_HEADS * SEQ * HEAD_DIM * 2);
    bf16_t* k_h     = (bf16_t*)alloc((size_t)BATCH * N_HEADS * SEQ * HEAD_DIM * 2);
    bf16_t* v_t     = (bf16_t*)alloc((size_t)BATCH * N_HEADS * HEAD_DIM * SEQ * 2);
    bf16_t* att_o   = (bf16_t*)alloc((size_t)M_TOT * D_MODEL * 2);

    // 1. precision conversion + weight transposes
    {
        int n4 = (M_TOT * D_MODEL) / 4;
        cvt_f32_to_bf16<<<(n4 + 255) / 256, 256, 0, stream>>>(x, xb, n4);
    }
    transpose_to_bf16<<<(D_MODEL * D_MODEL + 255) / 256, 256, 0, stream>>>(W_q,   WqT,   D_MODEL,   D_MODEL);
    transpose_to_bf16<<<(D_MODEL * KV_LATENT + 255) / 256, 256, 0, stream>>>(W_dkv, WdkvT, D_MODEL,   KV_LATENT);
    transpose_to_bf16<<<(KV_LATENT * D_MODEL + 255) / 256, 256, 0, stream>>>(W_uk,  WukT,  KV_LATENT, D_MODEL);
    transpose_to_bf16<<<(KV_LATENT * D_MODEL + 255) / 256, 256, 0, stream>>>(W_uv,  WuvT,  KV_LATENT, D_MODEL);
    transpose_to_bf16<<<(D_MODEL * D_MODEL + 255) / 256, 256, 0, stream>>>(W_o,   WoT,   D_MODEL,   D_MODEL);

    // 2. latent = x @ W_dkv  (f32 out), then LayerNorm -> bf16
    gemm_wmma<0><<<dim3(KV_LATENT / 64, M_TOT / 128), 256, 0, stream>>>(
        xb, WdkvT, lat_f32, KV_LATENT, D_MODEL, 1.f);
    ln_rows<<<(M_TOT * 32) / 256, 256, 0, stream>>>(lat_f32, ln_gamma, ln_beta, latn);

    // 3. q = x @ W_q  with fused RoPE + 1/sqrt(HEAD_DIM) scale -> [bh][t][hd]
    gemm_wmma<1><<<dim3(D_MODEL / 64, M_TOT / 128), 256, 0, stream>>>(
        xb, WqT, q_h, D_MODEL, D_MODEL, 0.125f);

    // 4. k = latn @ W_uk with fused RoPE -> [bh][t][hd]
    gemm_wmma<1><<<dim3(D_MODEL / 64, M_TOT / 128), 256, 0, stream>>>(
        latn, WukT, k_h, D_MODEL, KV_LATENT, 1.f);

    // 5. v = latn @ W_uv, stored transposed -> [bh][hd][t]
    gemm_wmma<2><<<dim3(D_MODEL / 64, M_TOT / 128), 256, 0, stream>>>(
        latn, WuvT, v_t, D_MODEL, KV_LATENT, 1.f);

    // 6. causal flash attention -> att_o bf16 (b,t,d)
    mla_attention<<<dim3(SEQ / 128, BATCH * N_HEADS), 256, 0, stream>>>(
        q_h, k_h, v_t, att_o);

    // 7. out = att_o @ W_o -> f32 d_out
    gemm_wmma<0><<<dim3(D_MODEL / 64, M_TOT / 128), 256, 0, stream>>>(
        att_o, WoT, (float*)d_out, D_MODEL, D_MODEL, 1.f);
}
